// EntEncoder_36472862278325
// MI455X (gfx1250) — compile-verified, matrix operand order checked by reading, not run on hardware
//
#include <hip/hip_runtime.h>
#include <hip/hip_fp16.h>
#include <math.h>

typedef __attribute__((ext_vector_type(16))) _Float16 v16h;
typedef __attribute__((ext_vector_type(8)))  _Float16 v8h;
typedef __attribute__((ext_vector_type(8)))  float    v8f;

#define HH 384
#define WW 384
#define PH 388            // padded height (+2 each side)
#define PW 392            // padded width  (+2 each side, +4 align)
#define C1 48             // hidden channels (NG*3)
#define C2 144            // final channels  (NG*3*NMIX)
#define NTAPS 13          // MASK_B active taps (rows -2,-1 full; row 0 cols -2..0)
#define KFLAT (NTAPS*C1)  // 624 flattened (tap,cin)
#define KCH 20            // ceil(624/32) K-chunks of 32

#if defined(__has_builtin)
#if __has_builtin(__builtin_amdgcn_global_load_async_to_lds_b128)
#define HAVE_ASYNC_LDS 1
#endif
#endif

#ifdef HAVE_ASYNC_LDS
// builtin expects: (int-vector16 AS1*, <lds ptr>, imm offset, cpol) per clang diagnostic
typedef int async_i4 __attribute__((__vector_size__(16)));
typedef __attribute__((address_space(1))) async_i4* g_as1_i4;
typedef __attribute__((address_space(3))) async_i4* l_as3_i4;
__device__ __forceinline__ void wait_async0() {
#if __has_builtin(__builtin_amdgcn_s_wait_asynccnt)
  __builtin_amdgcn_s_wait_asynccnt(0);
#else
  asm volatile("s_wait_asynccnt 0" ::: "memory");
#endif
}
#endif

// Activations are stored CHANNEL-FASTEST: act[(row*PW + col)*48 + ch] (f16).
// Each half of a WMMA B-fragment (8 consecutive K = 8 consecutive cin, never
// crossing a tap boundary since 48 % 8 == 0) is one aligned 16B LDS load.
// Furthermore the hi-lane run (K+8) is always the same tap at cin+8 (32kc mod 48
// and 32kc+16 mod 48 are never 40), so hi lanes use the SAME immediate offsets
// off a base advanced by 8 f16 -> zero per-chunk address math.

// ---- CDNA5 WMMA 16-bit fragment K index for (lane, f16-slot), ISA 7.12.2 ----
__device__ __forceinline__ int wmma_k_of(int lane, int s) {
  int k = (s < 8) ? s : s + 8;       // lanes 0-15: K = {0..7, 16..23}
  if (lane >= 16) k += 8;            // lanes 16-31: K = {8..15, 24..31}
  return k;
}

// compile-time foldable: LDS f16 offset (relative to (pxl+2)*48) for flattened K
// index kf, with LDS tile layout [3 rows][w cols][48 ch]. Min value is -96.
__device__ constexpr int frag_off(int kf, int w) {
  int tap = kf / C1, cin = kf % C1;
  int dy = 0, dx = 0;
  if (tap < 5)       { dy = -2; dx = tap - 2; }
  else if (tap < 10) { dy = -1; dx = tap - 7; }
  else               { dy =  0; dx = tap - 12; }
  return ((dy + 2) * w + dx) * C1 + cin;
}

// -------------------- zero workspace (halo padding must be zero each call) --------------------
__global__ void zero_kernel(uint4* __restrict__ p, long long n16) {
  long long i = (long long)blockIdx.x * blockDim.x + threadIdx.x;
  long long stride = (long long)gridDim.x * blockDim.x;
  uint4 z = {0u, 0u, 0u, 0u};
  for (; i < n16; i += stride) p[i] = z;
}

// -------------------- pack masked grouped weights into WMMA A-fragments --------------------
// wRes layout: [conv(10)][kc(20)][mtile(3)][lane(32)][slot(16)]  f16
// wFin layout: [kc(20)][mtile(9)][lane(32)][slot(16)]            f16
__global__ void prep_weights_kernel(const float* __restrict__ rw,
                                    const float* __restrict__ wf,
                                    _Float16* __restrict__ wRes,
                                    _Float16* __restrict__ wFin) {
  const int RES = 10 * KCH * 3 * 32 * 16;
  const int FIN = KCH * 9 * 32 * 16;
  int idx = blockIdx.x * 256 + threadIdx.x;
  if (idx < RES) {
    int t = idx;
    int s    = t & 15; t >>= 4;
    int lane = t & 31; t >>= 5;
    int mt   = t % 3;  t /= 3;
    int kc   = t % KCH; t /= KCH;
    int conv = t;                                  // 0..9
    int kf = kc * 32 + wmma_k_of(lane, s);
    float val = 0.f;
    if (kf < KFLAT) {
      int tap = kf / C1, cin = kf % C1;
      int m = mt * 16 + (lane & 15);
      if (cin / 3 == m / 3) {                      // block-diagonal group structure
        int dy = (tap < 5) ? -2 : (tap < 10 ? -1 : 0);
        int dx = (tap < 5) ? tap - 2 : (tap < 10 ? tap - 7 : tap - 12);
        // rw flat shape (5,2,48,3,5,5); conv = i*2+j
        val = rw[(((size_t)conv * C1 + m) * 3 + (cin % 3)) * 25 + (dy + 2) * 5 + (dx + 2)];
      }
    }
    wRes[idx] = (_Float16)val;
  } else if (idx < RES + FIN) {
    int t = idx - RES;
    int s    = t & 15; t >>= 4;
    int lane = t & 31; t >>= 5;
    int mt   = t % 9;  t /= 9;
    int kc   = t;
    int kf = kc * 32 + wmma_k_of(lane, s);
    float val = 0.f;
    if (kf < KFLAT) {
      int tap = kf / C1, cin = kf % C1;
      int m = mt * 16 + (lane & 15);               // 0..143
      if (cin / 3 == m / 9) {                      // group of cout m is m/9
        int dy = (tap < 5) ? -2 : (tap < 10 ? -1 : 0);
        int dx = (tap < 5) ? tap - 2 : (tap < 10 ? tap - 7 : tap - 12);
        val = wf[((size_t)m * 3 + (cin % 3)) * 25 + (dy + 2) * 5 + (dx + 2)];
      }
    }
    wFin[idx - RES] = (_Float16)val;
  }
}

// -------------------- layer 1: masked conv (16 groups, 1->3), VALU (cheap) --------------------
__global__ __launch_bounds__(256) void conv1_kernel(const int* __restrict__ data,
                                                    const float* __restrict__ w1,
                                                    const float* __restrict__ b1,
                                                    _Float16* __restrict__ out) {
  int idx = blockIdx.x * 256 + threadIdx.x;
  if (idx >= 16 * HH * WW) return;
  int g = idx / (HH * WW);
  int p = idx % (HH * WW);
  int y = p / WW, x = p % WW;
  float a0 = b1[g * 3 + 0], a1 = b1[g * 3 + 1], a2 = b1[g * 3 + 2];
  #pragma unroll
  for (int t = 0; t < 12; ++t) {                   // MASK_A: taps 0..11 (center excluded)
    int dy = (t < 5) ? -2 : (t < 10 ? -1 : 0);
    int dx = (t < 5) ? t - 2 : (t < 10 ? t - 7 : t - 12);
    int yy = y + dy, xx = x + dx;
    if (yy < 0 || xx < 0 || xx >= WW) continue;    // zero pad
    float v = (float)data[(size_t)g * HH * WW + (size_t)yy * WW + xx] - 3.5f;
    int wb = (dy + 2) * 5 + (dx + 2);
    a0 += v * w1[(size_t)(g * 3 + 0) * 25 + wb];
    a1 += v * w1[(size_t)(g * 3 + 1) * 25 + wb];
    a2 += v * w1[(size_t)(g * 3 + 2) * 25 + wb];
  }
  size_t o = ((size_t)(y + 2) * PW + (x + 2)) * C1 + (size_t)g * 3;  // channel-fastest
  out[o + 0] = (_Float16)fmaxf(a0, 0.f);
  out[o + 1] = (_Float16)fmaxf(a1, 0.f);
  out[o + 2] = (_Float16)fmaxf(a2, 0.f);
}

// -------------------- residual conv: WMMA implicit GEMM, 48ch, MASK_B --------------------
// 8 waves; tile = 128 pixels of one row; out = relu(W*in + b) [+ skip]
__global__ __launch_bounds__(256) void conv48_kernel(const _Float16* __restrict__ in,
                                                     _Float16* __restrict__ out,
                                                     const _Float16* __restrict__ skip,
                                                     const _Float16* __restrict__ wA,
                                                     const float* __restrict__ bias) {
  constexpr int TW  = 132;                         // tile cols (128 + 4 halo)
  constexpr int RW8 = TW * C1 / 8;                 // b128 chunks per row (792)
  __shared__ _Float16 sIn[3 * TW * C1];            // [3 rows][132 cols][48 ch]
  const int y   = blockIdx.y;
  const int x0  = blockIdx.x * 128;
  const int tid = threadIdx.x;

  // contiguous staging: 3 spans of 132*48 f16 (channel-fastest layout)
#ifdef HAVE_ASYNC_LDS
  for (int i = tid; i < 3 * RW8; i += 256) {
    int r = i / RW8, o = i % RW8;
    const uint4* src = (const uint4*)(in + ((size_t)(y + r) * PW + x0) * C1) + o;
    uint4* dst = (uint4*)sIn + (size_t)r * RW8 + o;
    __builtin_amdgcn_global_load_async_to_lds_b128((g_as1_i4)src, (l_as3_i4)dst, 0, 0);
  }
  wait_async0();
#else
  for (int i = tid; i < 3 * RW8; i += 256) {
    int r = i / RW8, o = i % RW8;
    const uint4* src = (const uint4*)(in + ((size_t)(y + r) * PW + x0) * C1) + o;
    *((uint4*)sIn + (size_t)r * RW8 + o) = *src;
  }
#endif
  if (tid < 48 && y + 3 < PH)                      // prefetch next row tile
    __builtin_prefetch(in + ((size_t)(y + 3) * PW + x0) * C1 + tid * 64, 0, 1);
  __syncthreads();

  const int lane = tid & 31;
  const bool hi  = (lane >= 16);
  const int wid  = tid >> 5;                       // N-tile 0..7
  const int pxl  = (wid << 4) + (lane & 15);       // local pixel 0..127
  // single base; all B loads below use compile-time immediate offsets (>= 0)
  const _Float16* bb = sIn + ((pxl + 2) * C1 - 96) + (hi ? 8 : 0);
  v8f acc0 = {}, acc1 = {}, acc2 = {};
  const v8h zero8 = {};

  #pragma unroll
  for (int kc = 0; kc < KCH; ++kc) {
    union { v16h v; v8h h[2]; } bu;
    bu.h[0] = *(const v8h*)(bb + (frag_off(kc * 32, TW) + 96));
    if (kc * 32 + 16 < KFLAT)
      bu.h[1] = *(const v8h*)(bb + (frag_off(kc * 32 + 16 < KFLAT ? kc * 32 + 16 : 0, TW) + 96));
    else
      bu.h[1] = zero8;                             // tail chunk: K >= 624 is zero
    v16h a0 = *(const v16h*)(wA + (size_t)((kc * 3 + 0) * 32 + lane) * 16);
    v16h a1 = *(const v16h*)(wA + (size_t)((kc * 3 + 1) * 32 + lane) * 16);
    v16h a2 = *(const v16h*)(wA + (size_t)((kc * 3 + 2) * 32 + lane) * 16);
    acc0 = __builtin_amdgcn_wmma_f32_16x16x32_f16(false, a0, false, bu.v, (short)0, acc0, false, false);
    acc1 = __builtin_amdgcn_wmma_f32_16x16x32_f16(false, a1, false, bu.v, (short)0, acc1, false, false);
    acc2 = __builtin_amdgcn_wmma_f32_16x16x32_f16(false, a2, false, bu.v, (short)0, acc2, false, false);
  }

  // D layout: lane<16 -> N=lane, M=vgpr; lane>=16 -> M=vgpr+8.
  // 8 accumulator elements = 8 consecutive channels at one pixel -> one 16B store.
  const int x     = x0 + pxl;
  const int mbase = hi ? 8 : 0;
  const size_t po = ((size_t)(y + 2) * PW + (x + 2)) * C1;
  v8f accs[3] = {acc0, acc1, acc2};
  #pragma unroll
  for (int mt = 0; mt < 3; ++mt) {
    const float4* bp = (const float4*)(bias + mt * 16 + mbase);
    float4 blo = bp[0], bhi = bp[1];
    float f[8];
    f[0] = accs[mt][0] + blo.x; f[1] = accs[mt][1] + blo.y;
    f[2] = accs[mt][2] + blo.z; f[3] = accs[mt][3] + blo.w;
    f[4] = accs[mt][4] + bhi.x; f[5] = accs[mt][5] + bhi.y;
    f[6] = accs[mt][6] + bhi.z; f[7] = accs[mt][7] + bhi.w;
    v8h o;
    if (skip) {
      v8h sk = *(const v8h*)(skip + po + mt * 16 + mbase);
      #pragma unroll
      for (int v = 0; v < 8; ++v) o[v] = (_Float16)(fmaxf(f[v], 0.f) + (float)sk[v]);
    } else {
      #pragma unroll
      for (int v = 0; v < 8; ++v) o[v] = (_Float16)fmaxf(f[v], 0.f);
    }
    *(v8h*)(out + po + mt * 16 + mbase) = o;
  }
}

// -------------------- final conv (144ch) fused with mixture CDF -> PMF --------------------
// 4 waves; tile = 64 pixels of one row
__global__ __launch_bounds__(128) void final_mixture_kernel(const _Float16* __restrict__ in,
                                                            const _Float16* __restrict__ wA,
                                                            const float* __restrict__ bf,
                                                            float* __restrict__ out) {
  constexpr int TW  = 68;                          // 64 + 4 halo
  constexpr int RW8 = TW * C1 / 8;                 // 408
  __shared__ _Float16 sIn[3 * TW * C1];            // [3][68][48]
  __shared__ float    sP[C2][64];                  // staged p for the 64-px tile
  const int y   = blockIdx.y;
  const int x0  = blockIdx.x * 64;
  const int tid = threadIdx.x;

#ifdef HAVE_ASYNC_LDS
  for (int i = tid; i < 3 * RW8; i += 128) {
    int r = i / RW8, o = i % RW8;
    const uint4* src = (const uint4*)(in + ((size_t)(y + r) * PW + x0) * C1) + o;
    uint4* dst = (uint4*)sIn + (size_t)r * RW8 + o;
    __builtin_amdgcn_global_load_async_to_lds_b128((g_as1_i4)src, (l_as3_i4)dst, 0, 0);
  }
  wait_async0();
#else
  for (int i = tid; i < 3 * RW8; i += 128) {
    int r = i / RW8, o = i % RW8;
    const uint4* src = (const uint4*)(in + ((size_t)(y + r) * PW + x0) * C1) + o;
    *((uint4*)sIn + (size_t)r * RW8 + o) = *src;
  }
#endif
  __syncthreads();

  const int lane = tid & 31;
  const bool hi  = (lane >= 16);
  const int wid  = tid >> 5;
  const int pxl  = (wid << 4) + (lane & 15);       // 0..63
  const _Float16* bb = sIn + ((pxl + 2) * C1 - 96) + (hi ? 8 : 0);
  v8f acc[9] = {};
  const v8h zero8 = {};

  #pragma unroll
  for (int kc = 0; kc < KCH; ++kc) {
    union { v16h v; v8h h[2]; } bu;
    bu.h[0] = *(const v8h*)(bb + (frag_off(kc * 32, TW) + 96));
    if (kc * 32 + 16 < KFLAT)
      bu.h[1] = *(const v8h*)(bb + (frag_off(kc * 32 + 16 < KFLAT ? kc * 32 + 16 : 0, TW) + 96));
    else
      bu.h[1] = zero8;
    #pragma unroll
    for (int mt = 0; mt < 9; ++mt) {
      v16h a = *(const v16h*)(wA + (size_t)((kc * 9 + mt) * 32 + lane) * 16);
      acc[mt] = __builtin_amdgcn_wmma_f32_16x16x32_f16(false, a, false, bu.v, (short)0, acc[mt], false, false);
    }
  }

  const int mbase = hi ? 8 : 0;
  #pragma unroll
  for (int mt = 0; mt < 9; ++mt)
    #pragma unroll
    for (int v = 0; v < 8; ++v) {
      int ch = mt * 16 + v + mbase;
      sP[ch][pxl] = acc[mt][v] + bf[ch];
    }
  __syncthreads();

  // mixture: 2 threads per pixel, 8 groups each
  const int pix = tid >> 1;
  const int g0  = (tid & 1) * 8;
  const int xg  = x0 + pix;
  const float inv_sqrt2 = 0.70710678118654752f;
  for (int g = g0; g < g0 + 8; ++g) {
    float l0 = sP[g * 9 + 0][pix], m0 = sP[g * 9 + 1][pix], s0 = sP[g * 9 + 2][pix];
    float l1 = sP[g * 9 + 3][pix], m1 = sP[g * 9 + 4][pix], s1 = sP[g * 9 + 5][pix];
    float l2 = sP[g * 9 + 6][pix], m2 = sP[g * 9 + 7][pix], s2 = sP[g * 9 + 8][pix];
    float mx = fmaxf(l0, fmaxf(l1, l2));
    float e0 = __expf(l0 - mx), e1 = __expf(l1 - mx), e2 = __expf(l2 - mx);
    float wn = 1.f / (e0 + e1 + e2);
    float w0 = e0 * wn, w1_ = e1 * wn, w2 = e2 * wn;
    float is0 = inv_sqrt2 / (log1pf(__expf(s0)) + 1e-6f);
    float is1 = inv_sqrt2 / (log1pf(__expf(s1)) + 1e-6f);
    float is2 = inv_sqrt2 / (log1pf(__expf(s2)) + 1e-6f);
    float cprev = 0.f;
    #pragma unroll
    for (int k = 0; k <= 8; ++k) {
      float e = (float)k - 4.0f;                   // edges - (BIAS+0.5)
      float c = w0 * 0.5f * (1.f + erff((e - m0) * is0))
              + w1_ * 0.5f * (1.f + erff((e - m1) * is1))
              + w2 * 0.5f * (1.f + erff((e - m2) * is2));
      if (k > 0)
        out[(((size_t)g * 8 + (k - 1)) * HH + y) * WW + xg] = (c - cprev) * 65536.0f;
      cprev = c;
    }
  }
}

// -------------------- host orchestration --------------------
extern "C" void kernel_launch(void* const* d_in, const int* in_sizes, int n_in,
                              void* d_out, int out_size, void* d_ws, size_t ws_size,
                              hipStream_t stream) {
  (void)in_sizes; (void)n_in; (void)out_size; (void)ws_size;
  const int*   data = (const int*)  d_in[0];
  const float* w1   = (const float*)d_in[1];
  const float* b1   = (const float*)d_in[2];
  const float* rw   = (const float*)d_in[3];
  const float* rb   = (const float*)d_in[4];
  const float* wf   = (const float*)d_in[5];
  const float* bf   = (const float*)d_in[6];
  float* out = (float*)d_out;

  const size_t bufElems = (size_t)PH * PW * C1;    // padded f16 activations, ch-fastest
  _Float16* bufA = (_Float16*)d_ws;
  _Float16* bufB = bufA + bufElems;
  _Float16* wRes = bufB + bufElems;                // [10][20][3][32][16]
  _Float16* wFin = wRes + (size_t)10 * KCH * 3 * 32 * 16;  // [20][9][32][16]

  // zero ping-pong buffers so conv halo padding reads 0 (required every call)
  long long n16 = (long long)(2 * bufElems * sizeof(_Float16)) / 16;
  zero_kernel<<<2048, 256, 0, stream>>>((uint4*)d_ws, n16);

  int prepJobs = 10 * KCH * 3 * 32 * 16 + KCH * 9 * 32 * 16;
  prep_weights_kernel<<<(prepJobs + 255) / 256, 256, 0, stream>>>(rw, wf, wRes, wFin);

  conv1_kernel<<<(16 * HH * WW + 255) / 256, 256, 0, stream>>>(data, w1, b1, bufA);

  dim3 cgrid(WW / 128, HH);
  for (int c = 0; c < 10; ++c) {                   // 5 residual blocks x 2 convs
    const _Float16* ip   = (c & 1) ? bufB : bufA;
    _Float16*       op   = (c & 1) ? bufA : bufB;
    const _Float16* skip = (c & 1) ? bufA : nullptr;   // h += relu(conv(t)) on 2nd conv
    conv48_kernel<<<cgrid, 256, 0, stream>>>(ip, op, skip,
                                             wRes + (size_t)c * KCH * 3 * 32 * 16,
                                             rb + c * C1);
  }

  dim3 fgrid(WW / 64, HH);
  final_mixture_kernel<<<fgrid, 128, 0, stream>>>(bufA, wFin, bf, out);
}
